// TripletLossWithGL_6296422056212
// MI455X (gfx1250) — matrix operand order repaired and checked by reading, not compiled, not run on hardware
//
#include <hip/hip_runtime.h>

typedef __attribute__((ext_vector_type(16))) _Float16 v16h;
typedef __attribute__((ext_vector_type(8)))  _Float16 v8h;
typedef __attribute__((ext_vector_type(8)))  float    v8f;

#define TL_N 8192
#define TL_D 128
#define TL_MARGIN 0.3f
#define SENT 3.0e38f
#define CGROUPS 4
#define ROWS_PER_BLOCK 128
#define TILES_PER_GROUP (TL_N / 16 / CGROUPS)   // 128 column tiles per block
#define LDS_STRIDE 136                          // halves per staged row (128 + 8 pad)

static __device__ __forceinline__ v16h tl_make_v16h(const _Float16* lo,
                                                    const _Float16* hi8) {
    v8h a = *(const v8h*)lo;
    v8h b = *(const v8h*)hi8;
    return __builtin_shufflevector(a, b, 0,1,2,3,4,5,6,7,8,9,10,11,12,13,14,15);
}

// ---------------------------------------------------------------------------
// Prep: fp32 -> fp16, row squared-norms from the f16-rounded values, and
// initialization of the per-row aggregate arrays + global accumulators.
// One block per row, 128 threads (4 waves of 32).
// ---------------------------------------------------------------------------
__global__ __launch_bounds__(128)
void tl_prep_kernel(const float* __restrict__ x,
                    _Float16* __restrict__ x16,
                    float* __restrict__ sq,
                    int* __restrict__ g_ap,
                    int* __restrict__ g_ag,
                    int* __restrict__ g_al,
                    float* __restrict__ accum) {
    const int row = blockIdx.x;
    const int t   = threadIdx.x;                 // 0..127
    float v = x[row * TL_D + t];
    _Float16 h = (_Float16)v;
    x16[row * TL_D + t] = h;
    float hv = (float)h;
    float p  = hv * hv;

    #pragma unroll
    for (int m = 16; m >= 1; m >>= 1) p += __shfl_xor(p, m, 32);

    __shared__ float red[4];
    if ((t & 31) == 0) red[t >> 5] = p;
    __syncthreads();
    if (t == 0) {
        sq[row]   = red[0] + red[1] + red[2] + red[3];
        g_ap[row] = 0;                           // float 0.0 (max over d2 >= 0)
        g_ag[row] = __float_as_int(SENT);        // min sentinel
        g_al[row] = __float_as_int(SENT);
    }
    if (row == 0 && t < 2) accum[t] = 0.0f;      // [0]=loss sum, [1]=valid cnt
}

// ---------------------------------------------------------------------------
// Fused Gram + triplet aggregation, LDS-staged B operands.
// Grid: (N/128) * CGROUPS blocks of 256 threads (8 waves).
//   - block owns 128 rows; wave w keeps its 16x128 A tile in registers.
//   - block iterates over its 128 column tiles; each 4KB B tile is staged
//     into LDS once (double-buffered) and consumed by all 8 waves.
//   - per-element work is in "e-space": e = sq_c - 2*gram (sqrt + sq_r
//     folded out by monotonicity), branchless cndmask selects.
//   - per-row partial aggregates combined across blocks with bit-ordered
//     integer atomic max/min on clamped non-negative floats.
// ---------------------------------------------------------------------------
__global__ __launch_bounds__(256)
void tl_gram_kernel(const _Float16* __restrict__ x16,
                    const float* __restrict__ sq,
                    const int* __restrict__ tgt,
                    int* __restrict__ g_ap,
                    int* __restrict__ g_ag,
                    int* __restrict__ g_al) {
    __shared__ _Float16 lds[2][16 * LDS_STRIDE];

    const int lane = threadIdx.x & 31;
    const int w    = threadIdx.x >> 5;           // wave id 0..7
    const int cg   = blockIdx.x & (CGROUPS - 1); // column group
    const int rb   = blockIdx.x / CGROUPS;       // row block
    const int r0   = rb * ROWS_PER_BLOCK + w * 16;
    const int n    = lane & 15;
    const int hi   = lane >> 4;

    // ---- A operand: rows r0..r0+15, K=128 as 4 chunks of 32 --------------
    // ISA 16-bit A 16x32 layout: lanes 0-15 carry K 0-7 / 16-23,
    // lanes 16-31 carry K 8-15 / 24-31 (m = lane&15).
    const int am  = r0 + n;
    const int akb = hi * 8;
    v16h A[4];
    #pragma unroll
    for (int kc = 0; kc < 4; ++kc) {
        const _Float16* p = x16 + am * TL_D + kc * 32 + akb;
        A[kc] = tl_make_v16h(p, p + 16);
    }

    // D-tile row metadata: VGPR v holds row m = v + 8*hi.
    float sqrow[8]; int trow[8];
    #pragma unroll
    for (int v = 0; v < 8; ++v) {
        int m = v + 8 * hi;
        sqrow[v] = sq[r0 + m];
        trow[v]  = tgt[r0 + m];
    }

    // Aggregates in e-space (e can be negative -> max init is -SENT).
    float ap[8], ag[8], al[8];
    #pragma unroll
    for (int v = 0; v < 8; ++v) { ap[v] = -SENT; ag[v] = SENT; al[v] = SENT; }

    // Staging coordinates: thread t moves 16B: row t>>4, halves (t&15)*8.
    const int srow = threadIdx.x >> 4;
    const int scol = (threadIdx.x & 15) * 8;
    const int ct0  = cg * TILES_PER_GROUP;

    // Prologue: stage first tile into buffer 0.
    *(v8h*)&lds[0][srow * LDS_STRIDE + scol] =
        *(const v8h*)(x16 + (ct0 * 16 + srow) * TL_D + scol);
    __syncthreads();

    for (int i = 0; i < TILES_PER_GROUP; ++i) {
        const int c0 = (ct0 + i) * 16;

        // Stage next tile into the other buffer.
        if (i + 1 < TILES_PER_GROUP) {
            *(v8h*)&lds[(i + 1) & 1][srow * LDS_STRIDE + scol] =
                *(const v8h*)(x16 + (c0 + 16 + srow) * TL_D + scol);
        }

        // B operand from LDS: lane column n, K hi*8 + {0-7, 16-23} (matches A).
        const _Float16* lb = &lds[i & 1][n * LDS_STRIDE + hi * 8];
        v8f acc = {};
        #pragma unroll
        for (int kc = 0; kc < 4; ++kc) {
            v16h B = tl_make_v16h(lb + kc * 32, lb + kc * 32 + 16);
            acc = __builtin_amdgcn_wmma_f32_16x16x32_f16(
                      false, A[kc], false, B, (short)0, acc, false, false);
        }

        const float sqc = sq[c0 + n];
        const int   tc  = tgt[c0 + n];
        #pragma unroll
        for (int v = 0; v < 8; ++v) {
            float e  = __builtin_fmaf(-2.0f, acc[v], sqc);  // d2 - sqrow[v]
            int   tr = trow[v];
            ap[v] = fmaxf(ap[v], (tc == tr) ? e : -SENT);
            ag[v] = fminf(ag[v], (tc >  tr) ? e :  SENT);
            al[v] = fminf(al[v], (tc <  tr) ? e :  SENT);
        }
        __syncthreads();
    }

    // ---- reduce across the 16 n-lanes (within each lane half) ------------
    #pragma unroll
    for (int mask = 1; mask < 16; mask <<= 1) {
        #pragma unroll
        for (int v = 0; v < 8; ++v) {
            ap[v] = fmaxf(ap[v], __shfl_xor(ap[v], mask, 32));
            ag[v] = fminf(ag[v], __shfl_xor(ag[v], mask, 32));
            al[v] = fminf(al[v], __shfl_xor(al[v], mask, 32));
        }
    }

    // ---- combine across blocks: bit-ordered int atomics on d2 >= 0 -------
    if (n == 0) {
        #pragma unroll
        for (int v = 0; v < 8; ++v) {
            const int r = r0 + v + 8 * hi;
            float apd = fmaxf(ap[v] + sqrow[v], 0.0f);   // back to d2 space
            float agd = fmaxf(ag[v] + sqrow[v], 0.0f);   // SENT + ~250 == SENT
            float ald = fmaxf(al[v] + sqrow[v], 0.0f);
            atomicMax(&g_ap[r], __float_as_int(apd));
            atomicMin(&g_ag[r], __float_as_int(agd));
            atomicMin(&g_al[r], __float_as_int(ald));
        }
    }
}

// ---------------------------------------------------------------------------
// Per-row loss from the combined aggregates (sqrt happens here, once/row),
// block-reduced before the global atomicAdd.
// ---------------------------------------------------------------------------
__global__ __launch_bounds__(256)
void tl_row_kernel(const int* __restrict__ g_ap,
                   const int* __restrict__ g_ag,
                   const int* __restrict__ g_al,
                   float* __restrict__ accum) {
    const int r = blockIdx.x * 256 + threadIdx.x;
    const float apd = __int_as_float(g_ap[r]);
    const float agd = __int_as_float(g_ag[r]);
    const float ald = __int_as_float(g_al[r]);

    const float dist_ap = sqrtf(fmaxf(apd, 1e-12f));
    const bool hg = agd < 1.0e37f;
    const bool hl = ald < 1.0e37f;
    const float an_g = hg ? sqrtf(fmaxf(agd, 1e-12f)) : 0.0f;
    const float an_l = hl ? sqrtf(fmaxf(ald, 1e-12f)) : 0.0f;
    const bool valid = hg && hl;
    float contrib = valid ? fmaxf(dist_ap - fabsf(an_g - an_l) + TL_MARGIN, 0.0f)
                          : 0.0f;
    float cnt = valid ? 1.0f : 0.0f;

    #pragma unroll
    for (int mask = 1; mask < 32; mask <<= 1) {
        contrib += __shfl_xor(contrib, mask, 32);
        cnt     += __shfl_xor(cnt,     mask, 32);
    }
    __shared__ float sc[8], sn[8];
    if ((threadIdx.x & 31) == 0) {
        sc[threadIdx.x >> 5] = contrib;
        sn[threadIdx.x >> 5] = cnt;
    }
    __syncthreads();
    if (threadIdx.x == 0) {
        float s = 0.f, c = 0.f;
        #pragma unroll
        for (int ww = 0; ww < 8; ++ww) { s += sc[ww]; c += sn[ww]; }
        atomicAdd(&accum[0], s);
        atomicAdd(&accum[1], c);
    }
}

__global__ void tl_finalize_kernel(const float* __restrict__ accum,
                                   float* __restrict__ out) {
    out[0] = accum[0] / fmaxf(accum[1], 1.0f);
}

// ---------------------------------------------------------------------------
extern "C" void kernel_launch(void* const* d_in, const int* in_sizes, int n_in,
                              void* d_out, int out_size, void* d_ws, size_t ws_size,
                              hipStream_t stream) {
    const float* x   = (const float*)d_in[0];   // inputs  (8192 x 128) fp32
    const int*   tgt = (const int*)d_in[1];     // targets (8192) int32

    char* ws = (char*)d_ws;
    size_t off = 0;
    _Float16* x16 = (_Float16*)(ws + off); off += (size_t)TL_N * TL_D * 2;  // 2 MB
    float* sq     = (float*)(ws + off);    off += (size_t)TL_N * 4;
    int*   g_ap   = (int*)(ws + off);      off += (size_t)TL_N * 4;
    int*   g_ag   = (int*)(ws + off);      off += (size_t)TL_N * 4;
    int*   g_al   = (int*)(ws + off);      off += (size_t)TL_N * 4;
    float* accum  = (float*)(ws + off);

    tl_prep_kernel<<<TL_N, 128, 0, stream>>>(x, x16, sq, g_ap, g_ag, g_al, accum);
    tl_gram_kernel<<<(TL_N / ROWS_PER_BLOCK) * CGROUPS, 256, 0, stream>>>(
        x16, sq, tgt, g_ap, g_ag, g_al);
    tl_row_kernel<<<TL_N / 256, 256, 0, stream>>>(g_ap, g_ag, g_al, accum);
    tl_finalize_kernel<<<1, 1, 0, stream>>>(accum, (float*)d_out);
}
